// GatedGCNNet_68101001445812
// MI455X (gfx1250) — matrix operand order repaired and checked by reading, not compiled, not run on hardware
//
#include <hip/hip_runtime.h>
#include <hip/hip_bf16.h>
#include <math.h>

typedef __attribute__((ext_vector_type(16))) _Float16 v16h;
typedef __attribute__((ext_vector_type(8)))  float    v8f;
typedef __attribute__((ext_vector_type(4)))  unsigned int u32x4;
typedef __attribute__((ext_vector_type(8)))  int      i32x8;
typedef __attribute__((ext_vector_type(4)))  int      i32x4;

#define NNODES 30000
#define NEDGES 720000

// ---------------------------------------------------------------------------
// TDM: 1D copy of nelem f16 elements global -> LDS via tensor_load_to_lds.
// D# per CDNA5 ISA ch.8: group0 {count=1, lds_addr, global_addr, type=2},
// group1 {data_size=2B, tensor_dim0=nelem, tensor_dim1=1, tile_dim0=nelem,
//         tile_dim1=1, dim0_stride=nelem}. Issue from one wave only
// (TDM ignores EXEC), wait TENSORcnt, then barrier.
// This toolchain exposes the 6-arg builtin (extra i32x8 group before cpol).
// ---------------------------------------------------------------------------
__device__ __forceinline__ void tdm_load_1d_f16(const _Float16* gsrc,
                                                _Float16* ldst,
                                                unsigned int nelem)
{
    const unsigned long long ga = (unsigned long long)(size_t)gsrc;
    const unsigned int lo = (unsigned int)(size_t)(void*)ldst; // LDS byte offset
    u32x4 g0;
    g0[0] = 1u;                                    // count=1
    g0[1] = lo;                                    // lds_addr
    g0[2] = (unsigned int)ga;                      // global_addr[31:0]
    g0[3] = (unsigned int)((ga >> 32) & 0x1ffffffull) | 0x80000000u; // [56:32] | type=2
    i32x8 g1;
    g1[0] = (int)(1u << 16);                       // data_size = 2 bytes
    g1[1] = (int)((nelem & 0xffffu) << 16);        // tensor_dim0[15:0]
    g1[2] = (int)(((nelem >> 16) & 0xffffu) | (1u << 16)); // dim0[31:16], dim1=1
    g1[3] = (int)((nelem & 0xffffu) << 16);        // tile_dim0
    g1[4] = 1;                                     // tile_dim1=1, tile_dim2=0
    g1[5] = (int)nelem;                            // tensor_dim0_stride[31:0]
    g1[6] = 0;
    g1[7] = 0;
    const i32x4 z4 = (i32x4)0;
    const i32x8 z8 = (i32x8)0;
    __builtin_amdgcn_tensor_load_to_lds(g0, g1, z4, z4, z8, 0);
}

// ---------------------------------------------------------------------------
// Generic WMMA linear: out[nrows,64] = in[nrows,K] @ W16[64,K]^T + b
// W16 is pre-converted f16, staged to LDS by the Tensor Data Mover.
// One wave = 16-row tile; block = 8 waves = 128 rows.
// ---------------------------------------------------------------------------
__global__ __launch_bounds__(256) void linear_wmma_kernel(
    const float* __restrict__ in, const _Float16* __restrict__ W16,
    const float* __restrict__ bias, float* __restrict__ out,
    int nrows, int K, int lda, int ldc)
{
    __shared__ _Float16 Wl[64 * 320];
    __shared__ float bl[64];
    const int tid = threadIdx.x;
    if (tid < 32) {                       // wave 0 issues the TDM op
        tdm_load_1d_f16(W16, &Wl[0], (unsigned int)(64 * K));
        __builtin_amdgcn_s_wait_tensorcnt(0);
    }
    if (tid >= 64 && tid < 128) bl[tid - 64] = bias[tid - 64];
    __syncthreads();

    const int wave = tid >> 5, lane = tid & 31;
    const int rowbase = (blockIdx.x * 8 + wave) * 16;
    if (rowbase >= nrows) return;

    const int n    = lane & 15;   // A: row-in-tile; B/D: column-in-tile
    const int half = lane >> 4;
    const int kab  = half * 16;   // A-fragment K offset per lane half
    const int kbb  = half * 16;   // B-fragment K offset per lane half
    int arow = rowbase + n; if (arow >= nrows) arow = nrows - 1;
    const float* arp = in + (size_t)arow * lda;

    v8f acc[4];
#pragma unroll
    for (int j = 0; j < 4; ++j) {
        const float bv = bl[j * 16 + n];
#pragma unroll
        for (int r = 0; r < 8; ++r) acc[j][r] = bv;
    }

    for (int kt = 0; kt < K; kt += 32) {
        v16h a;
#pragma unroll
        for (int p = 0; p < 8; ++p) {
            const int k = kt + 2 * p + kab;
            const float2 xv = *(const float2*)(arp + k);
            a[2 * p]     = (_Float16)xv.x;
            a[2 * p + 1] = (_Float16)xv.y;
        }
#pragma unroll
        for (int j = 0; j < 4; ++j) {
            v16h b;
            const _Float16* wp = &Wl[(j * 16 + n) * K + kt + kbb];
#pragma unroll
            for (int e = 0; e < 16; ++e) b[e] = wp[e];
            acc[j] = __builtin_amdgcn_wmma_f32_16x16x32_f16(
                false, a, false, b, (short)0, acc[j], false, false);
        }
    }

    const int mb = half * 8;
    if (rowbase + 16 <= nrows) {          // fast path: whole tile in range
        float* orp = out + (size_t)(rowbase + mb) * ldc + n;
#pragma unroll
        for (int j = 0; j < 4; ++j)
#pragma unroll
            for (int r = 0; r < 8; ++r)
                orp[(size_t)r * ldc + j * 16] = acc[j][r];
    } else {
#pragma unroll
        for (int j = 0; j < 4; ++j)
#pragma unroll
            for (int r = 0; r < 8; ++r) {
                const int row = rowbase + mb + r;
                if (row < nrows) out[(size_t)row * ldc + j * 16 + n] = acc[j][r];
            }
    }
}

// ---------------------------------------------------------------------------
// LSTM (one direction). Wave owns 16 nodes; gates [16,256] per timestep via
// WMMA. h,c live in D-fragment registers; h goes through a per-wave LDS tile
// to rebuild the A-fragment each step (same-wave DS ops are in-order).
// w16 = pre-converted f16 Wih||Whh (contiguous), TDM-staged to LDS.
// ---------------------------------------------------------------------------
__global__ __launch_bounds__(256) void lstm_kernel(
    const int* __restrict__ text, const int* __restrict__ tlen,
    const float* __restrict__ emb,
    const _Float16* __restrict__ w16,
    const float* __restrict__ bih, const float* __restrict__ bhh,
    float* __restrict__ hout, int nNodes, int reverse)
{
    __shared__ _Float16 Wb[2 * 256 * 64];
    __shared__ float bl[256];
    __shared__ float htr[8][16][64];
    const _Float16* Wi = Wb;
    const _Float16* Wh = Wb + 256 * 64;

    const int tid = threadIdx.x;
    if (tid < 32) {
        tdm_load_1d_f16(w16, &Wb[0], 2u * 256u * 64u);
        __builtin_amdgcn_s_wait_tensorcnt(0);
    }
    if (tid >= 32 && tid < 32 + 256) bl[tid - 32] = bih[tid - 32] + bhh[tid - 32];
    {
        float* hz = &htr[0][0][0];
        for (int i = tid; i < 8 * 16 * 64; i += 256) hz[i] = 0.0f;
    }
    __syncthreads();

    const int wave = tid >> 5, lane = tid & 31;
    const int n = lane & 15, half = lane >> 4;
    const int kab = half * 16, kbb = half * 16, mb = half * 8;
    const int base = blockIdx.x * 128 + wave * 16;

    int nA = base + n; if (nA >= nNodes) nA = nNodes - 1;
    const int  lenA = tlen[nA];
    const int* trow = text + (size_t)nA * 16;

    int lenD[8];
#pragma unroll
    for (int r = 0; r < 8; ++r) {
        const int nd = base + mb + r;
        lenD[r] = (nd < nNodes) ? tlen[nd] : 0;
    }

    float cD[4][8], hD[4][8];
#pragma unroll
    for (int j = 0; j < 4; ++j)
#pragma unroll
        for (int r = 0; r < 8; ++r) { cD[j][r] = 0.0f; hD[j][r] = 0.0f; }

    for (int t = 0; t < 16; ++t) {
        int ti = reverse ? (lenA - 1 - t) : t;
        if (ti < 0) ti = 0;
        const int tok = trow[ti];
        const float* xr = emb + (size_t)tok * 64;

        v16h xA[2], hA[2];
#pragma unroll
        for (int q = 0; q < 2; ++q)
#pragma unroll
            for (int p = 0; p < 8; ++p) {
                const int k = q * 32 + 2 * p + kab;
                const float2 xv = *(const float2*)(xr + k);
                xA[q][2 * p]     = (_Float16)xv.x;
                xA[q][2 * p + 1] = (_Float16)xv.y;
                const float2 hv = *(const float2*)(&htr[wave][n][k]);
                hA[q][2 * p]     = (_Float16)hv.x;
                hA[q][2 * p + 1] = (_Float16)hv.y;
            }

#pragma unroll
        for (int j = 0; j < 4; ++j) {           // 16 gate columns per pass
            v8f g[4];                           // i, f, g, o tiles
#pragma unroll
            for (int gt = 0; gt < 4; ++gt) {
                const float bv = bl[gt * 64 + j * 16 + n];
#pragma unroll
                for (int r = 0; r < 8; ++r) g[gt][r] = bv;
            }
#pragma unroll
            for (int q = 0; q < 2; ++q) {
#pragma unroll
                for (int gt = 0; gt < 4; ++gt) {
                    const int oc = gt * 64 + j * 16 + n;
                    v16h bi, bh;
                    const _Float16* wpi = &Wi[oc * 64 + q * 32 + kbb];
                    const _Float16* wph = &Wh[oc * 64 + q * 32 + kbb];
#pragma unroll
                    for (int e = 0; e < 16; ++e) { bi[e] = wpi[e]; bh[e] = wph[e]; }
                    g[gt] = __builtin_amdgcn_wmma_f32_16x16x32_f16(
                        false, xA[q], false, bi, (short)0, g[gt], false, false);
                    g[gt] = __builtin_amdgcn_wmma_f32_16x16x32_f16(
                        false, hA[q], false, bh, (short)0, g[gt], false, false);
                }
            }
#pragma unroll
            for (int r = 0; r < 8; ++r) {
                const float iv = 1.0f / (1.0f + __expf(-g[0][r]));
                const float fv = 1.0f / (1.0f + __expf(-g[1][r]));
                const float gv = tanhf(g[2][r]);
                const float ov = 1.0f / (1.0f + __expf(-g[3][r]));
                const float cn = fv * cD[j][r] + iv * gv;
                const float hn = ov * tanhf(cn);
                if (t < lenD[r]) { cD[j][r] = cn; hD[j][r] = hn; }
                htr[wave][mb + r][j * 16 + n] = hD[j][r];
            }
        }
    }

#pragma unroll
    for (int j = 0; j < 4; ++j)
#pragma unroll
        for (int r = 0; r < 8; ++r) {
            const int nd = base + mb + r;
            if (nd < nNodes) hout[(size_t)nd * 64 + j * 16 + n] = hD[j][r];
        }
}

// ---------------------------------------------------------------------------
__global__ void f32_to_f16_kernel(const float* __restrict__ src,
                                  _Float16* __restrict__ dst, int n)
{
    const int i = blockIdx.x * 256 + threadIdx.x;
    if (i < n) dst[i] = (_Float16)src[i];
}

__global__ void embed_h_kernel(const float* __restrict__ hin,
                               const float* __restrict__ W,
                               const float* __restrict__ b,
                               float* __restrict__ out, int n)
{
    const int idx = blockIdx.x * 256 + threadIdx.x;
    if (idx >= n * 64) return;
    const int node = idx >> 6, col = idx & 63;
    float acc = b[col];
#pragma unroll
    for (int k = 0; k < 8; ++k) acc += hin[(size_t)node * 8 + k] * W[col * 8 + k];
    out[idx] = acc;
}

__global__ void embed_e_kernel(const float* __restrict__ ein,
                               const float* __restrict__ W,
                               const float* __restrict__ b,
                               float* __restrict__ out, int n)
{
    const int idx = blockIdx.x * 256 + threadIdx.x;
    if (idx >= n * 64) return;
    const int eI = idx >> 6, col = idx & 63;
    float acc = b[col] + ein[(size_t)eI * 2] * W[col * 2]
                       + ein[(size_t)eI * 2 + 1] * W[col * 2 + 1];
    out[idx] = acc;
}

// temb = normalize(0.5*(hf+hb)); hh = hemb + temb  -> hcur and allH block 0
__global__ __launch_bounds__(256) void temb_kernel(
    const float* __restrict__ hf, const float* __restrict__ hb,
    const float* __restrict__ hemb, float* __restrict__ hcur,
    float* __restrict__ allH, int nNodes)
{
    const int node = blockIdx.x * 8 + (threadIdx.x >> 5);
    const int lane = threadIdx.x & 31;
    if (node >= nNodes) return;
    const size_t o = (size_t)node * 64;
    const float t0 = 0.5f * (hf[o + lane] + hb[o + lane]);
    const float t1 = 0.5f * (hf[o + 32 + lane] + hb[o + 32 + lane]);
    float ss = t0 * t0 + t1 * t1;
#pragma unroll
    for (int off = 16; off > 0; off >>= 1) ss += __shfl_xor(ss, off, 32);
    const float inv = 1.0f / fmaxf(sqrtf(ss), 1e-12f);
    const float v0 = hemb[o + lane] + t0 * inv;
    const float v1 = hemb[o + 32 + lane] + t1 * inv;
    hcur[o + lane] = v0;      hcur[o + 32 + lane] = v1;
    allH[(size_t)node * 320 + lane] = v0;
    allH[(size_t)node * 320 + 32 + lane] = v1;
}

__global__ void fill_zero_kernel(float* p, int n)
{
    const int i = blockIdx.x * 256 + threadIdx.x;
    if (i < n) p[i] = 0.0f;
}

// e_new = Ce + Dh[src] + Eh[dst]; sigma = sigmoid(e_new);
// num[dst]+=sigma*Bh[src]; den[dst]+=sigma; e += relu(e_new*snorm_e) in place
__global__ void edge_combine_kernel(
    const float* __restrict__ Ce, const float* __restrict__ Dh,
    const float* __restrict__ Eh, const float* __restrict__ Bh,
    const int* __restrict__ src, const int* __restrict__ dst,
    const float* __restrict__ sne, float* __restrict__ ecur,
    float* __restrict__ num, float* __restrict__ den, int nE)
{
    const int idx = blockIdx.x * 256 + threadIdx.x;
    if (idx >= nE * 64) return;
    const int eI = idx >> 6, j = idx & 63;
    const int s = src[eI], d = dst[eI];
    const float en = Ce[idx] + Dh[(size_t)s * 64 + j] + Eh[(size_t)d * 64 + j];
    const float sg = 1.0f / (1.0f + __expf(-en));
    atomicAdd(&num[(size_t)d * 64 + j], sg * Bh[(size_t)s * 64 + j]);
    atomicAdd(&den[(size_t)d * 64 + j], sg);
    ecur[idx] += fmaxf(en * sne[eI], 0.0f);
}

// h1 = h_in + relu((Ah + num/(den+1e-6)) * snorm_n) -> allH block (layer+1)
__global__ void node_finalize_kernel(
    const float* __restrict__ hcur, const float* __restrict__ Ah,
    const float* __restrict__ num, const float* __restrict__ den,
    const float* __restrict__ snn, float* __restrict__ allH,
    int layer, int nNodes)
{
    const int idx = blockIdx.x * 256 + threadIdx.x;
    if (idx >= nNodes * 64) return;
    const int node = idx >> 6, j = idx & 63;
    float hn = Ah[idx] + num[idx] / (den[idx] + 1e-6f);
    hn = fmaxf(hn * snn[node], 0.0f);
    allH[(size_t)node * 320 + (layer + 1) * 64 + j] = hcur[idx] + hn;
}

// row-wise LayerNorm + relu over first w columns of allH -> tmp (stride 320)
__global__ __launch_bounds__(256) void ln_relu_kernel(
    const float* __restrict__ allH, const float* __restrict__ g,
    const float* __restrict__ bb, float* __restrict__ tmp, int nNodes, int w)
{
    const int node = blockIdx.x * 8 + (threadIdx.x >> 5);
    const int lane = threadIdx.x & 31;
    if (node >= nNodes) return;
    const float* row = allH + (size_t)node * 320;
    float s = 0.0f, s2 = 0.0f;
    for (int k = lane; k < w; k += 32) {
        const float v = row[k];
        s += v; s2 += v * v;
    }
#pragma unroll
    for (int off = 16; off > 0; off >>= 1) {
        s  += __shfl_xor(s, off, 32);
        s2 += __shfl_xor(s2, off, 32);
    }
    const float mu  = s / (float)w;
    const float var = s2 / (float)w - mu * mu;
    const float rs  = rsqrtf(var + 1e-5f);
    for (int k = lane; k < w; k += 32) {
        const float v = (row[k] - mu) * rs * g[k] + bb[k];
        tmp[(size_t)node * 320 + k] = fmaxf(v, 0.0f);
    }
}

// MLP head 64 -> 32 -> 16 -> 5 (tiny; VALU, weights hit scalar cache)
__global__ __launch_bounds__(256) void mlp_kernel(
    const float* __restrict__ hin,
    const float* __restrict__ w0, const float* __restrict__ b0,
    const float* __restrict__ w1, const float* __restrict__ b1,
    const float* __restrict__ w2, const float* __restrict__ b2,
    float* __restrict__ out, int nNodes)
{
    const int node = blockIdx.x * 256 + threadIdx.x;
    if (node >= nNodes) return;
    float y1[32];
#pragma unroll
    for (int o = 0; o < 32; ++o) y1[o] = b0[o];
    for (int k = 0; k < 64; ++k) {
        const float xv = hin[(size_t)node * 64 + k];
#pragma unroll
        for (int o = 0; o < 32; ++o) y1[o] += xv * w0[o * 64 + k];
    }
#pragma unroll
    for (int o = 0; o < 32; ++o) y1[o] = fmaxf(y1[o], 0.0f);

    float y2[16];
#pragma unroll
    for (int o = 0; o < 16; ++o) y2[o] = b1[o];
#pragma unroll
    for (int k = 0; k < 32; ++k) {
        const float xv = y1[k];
#pragma unroll
        for (int o = 0; o < 16; ++o) y2[o] += xv * w1[o * 32 + k];
    }
#pragma unroll
    for (int o = 0; o < 16; ++o) y2[o] = fmaxf(y2[o], 0.0f);

    float y3[5];
#pragma unroll
    for (int o = 0; o < 5; ++o) y3[o] = b2[o];
#pragma unroll
    for (int k = 0; k < 16; ++k) {
        const float xv = y2[k];
#pragma unroll
        for (int o = 0; o < 5; ++o) y3[o] += xv * w2[o * 16 + k];
    }
#pragma unroll
    for (int o = 0; o < 5; ++o) out[(size_t)node * 5 + o] = y3[o];
}

// ---------------------------------------------------------------------------
extern "C" void kernel_launch(void* const* d_in, const int* in_sizes, int n_in,
                              void* d_out, int out_size, void* d_ws, size_t ws_size,
                              hipStream_t stream)
{
    const int NN = NNODES, NE = NEDGES;
    const int*   src  = (const int*)d_in[0];
    const int*   dst  = (const int*)d_in[1];
    const float* h0   = (const float*)d_in[2];
    const float* e0   = (const float*)d_in[3];
    const int*   text = (const int*)d_in[4];
    const int*   tlen = (const int*)d_in[5];
    const float* snn  = (const float*)d_in[6];
    const float* sne  = (const float*)d_in[7];

    int pi = 8;
    auto F = [&]() -> const float* { return (const float*)d_in[pi++]; };
    const float* emb_text = F();
    const float* ehw = F(); const float* ehb = F();
    const float* eew = F(); const float* eeb = F();
    const float* lf_wih = F(); const float* lf_whh = F();
    const float* lf_bih = F(); const float* lf_bhh = F();
    const float* lb_wih = F(); const float* lb_whh = F();
    const float* lb_bih = F(); const float* lb_bhh = F();
    const float* Lw[4][5]; const float* Lb[4][5];          // A,B,C,D,E
    for (int i = 0; i < 4; ++i)
        for (int j = 0; j < 5; ++j) { Lw[i][j] = F(); Lb[i][j] = F(); }
    const float* Dg[4]; const float* Dbb[4]; const float* Dlw[4]; const float* Dlb[4];
    for (int i = 0; i < 4; ++i) { Dg[i] = F(); Dbb[i] = F(); Dlw[i] = F(); Dlb[i] = F(); }
    const float* m0w = F(); const float* m0b = F();
    const float* m1w = F(); const float* m1b = F();
    const float* m2w = F(); const float* m2b = F();

    float* p = (float*)d_ws;
    auto alloc = [&](size_t nf) { float* q = p; p += nf; return q; };
    float* hemb = alloc((size_t)NN * 64);
    float* hf   = alloc((size_t)NN * 64);
    float* hb   = alloc((size_t)NN * 64);
    float* hcur = alloc((size_t)NN * 64);
    float* allH = alloc((size_t)NN * 320);
    float* Ah   = alloc((size_t)NN * 64);
    float* Bh   = alloc((size_t)NN * 64);
    float* Dh   = alloc((size_t)NN * 64);
    float* Eh   = alloc((size_t)NN * 64);
    float* num  = alloc((size_t)NN * 64);
    float* den  = alloc((size_t)NN * 64);   // contiguous after num
    float* tmp  = alloc((size_t)NN * 320);
    float* ecur = alloc((size_t)NE * 64);
    float* Ce   = alloc((size_t)NE * 64);

    const dim3 B(256);

    // --- pre-convert all GEMM/LSTM weights to f16 (contiguous arena in ws) ---
    _Float16* arena = (_Float16*)p;
    size_t aoff = 0;
    auto cvt = [&](const float* s, int n) -> const _Float16* {
        _Float16* dp = arena + aoff; aoff += (size_t)n;
        f32_to_f16_kernel<<<(n + 255) / 256, B, 0, stream>>>(s, dp, n);
        return dp;
    };
    const _Float16* lf16 = cvt(lf_wih, 256 * 64);  cvt(lf_whh, 256 * 64); // contiguous pair
    const _Float16* lb16 = cvt(lb_wih, 256 * 64);  cvt(lb_whh, 256 * 64); // contiguous pair
    const _Float16* Lw16[4][5];
    for (int i = 0; i < 4; ++i)
        for (int j = 0; j < 5; ++j) Lw16[i][j] = cvt(Lw[i][j], 64 * 64);
    const _Float16* Dlw16[4];
    for (int i = 0; i < 4; ++i) Dlw16[i] = cvt(Dlw[i], 64 * (i + 2) * 64);

    embed_h_kernel<<<(NN * 64 + 255) / 256, B, 0, stream>>>(h0, ehw, ehb, hemb, NN);
    embed_e_kernel<<<(NE * 64 + 255) / 256, B, 0, stream>>>(e0, eew, eeb, ecur, NE);

    const int gL = (NN + 127) / 128;
    lstm_kernel<<<gL, B, 0, stream>>>(text, tlen, emb_text,
                                      lf16, lf_bih, lf_bhh, hf, NN, 0);
    lstm_kernel<<<gL, B, 0, stream>>>(text, tlen, emb_text,
                                      lb16, lb_bih, lb_bhh, hb, NN, 1);
    temb_kernel<<<(NN + 7) / 8, B, 0, stream>>>(hf, hb, hemb, hcur, allH, NN);

    const int gNode = (NN + 127) / 128;
    const int gEdge = (NE + 127) / 128;
    for (int l = 0; l < 4; ++l) {
        linear_wmma_kernel<<<gNode, B, 0, stream>>>(hcur, Lw16[l][0], Lb[l][0], Ah, NN, 64, 64, 64);
        linear_wmma_kernel<<<gNode, B, 0, stream>>>(hcur, Lw16[l][1], Lb[l][1], Bh, NN, 64, 64, 64);
        linear_wmma_kernel<<<gNode, B, 0, stream>>>(hcur, Lw16[l][3], Lb[l][3], Dh, NN, 64, 64, 64);
        linear_wmma_kernel<<<gNode, B, 0, stream>>>(hcur, Lw16[l][4], Lb[l][4], Eh, NN, 64, 64, 64);
        linear_wmma_kernel<<<gEdge, B, 0, stream>>>(ecur, Lw16[l][2], Lb[l][2], Ce, NE, 64, 64, 64);
        fill_zero_kernel<<<(NN * 128 + 255) / 256, B, 0, stream>>>(num, NN * 128);
        edge_combine_kernel<<<(NE * 64 + 255) / 256, B, 0, stream>>>(
            Ce, Dh, Eh, Bh, src, dst, sne, ecur, num, den, NE);
        node_finalize_kernel<<<(NN * 64 + 255) / 256, B, 0, stream>>>(
            hcur, Ah, num, den, snn, allH, l, NN);
        const int w = (l + 2) * 64;
        ln_relu_kernel<<<(NN + 7) / 8, B, 0, stream>>>(allH, Dg[l], Dbb[l], tmp, NN, w);
        linear_wmma_kernel<<<gNode, B, 0, stream>>>(tmp, Dlw16[l], Dlb[l], hcur, NN, w, 320, 64);
    }

    mlp_kernel<<<(NN + 255) / 256, B, 0, stream>>>(
        hcur, m0w, m0b, m1w, m1b, m2w, m2b, (float*)d_out, NN);
}
